// EuclideanCodebook_5471788335793
// MI455X (gfx1250) — matrix operand order repaired and checked by reading, not compiled, not run on hardware
//
#include <hip/hip_runtime.h>
#include <hip/hip_bf16.h>

typedef __attribute__((ext_vector_type(16))) __bf16 v16bf;
typedef __attribute__((ext_vector_type(8)))  float  v8f;

#define N_CODES  4096
#define DIM      1280
#define N_TOKENS 32768
#define KCHUNKS  (DIM / 32)      // 40 k-chunks of 32
#define NTILES   (N_CODES / 16)  // 256 code tiles
#define MTILES   (N_TOKENS / 16) // 2048 token tiles
#define WAVES_PER_WG 4
#define NBLK     4               // N-tile register blocking per wave

// ---------------------------------------------------------------------------
// Kernel 1: convert embed (fp32 row-major) into bf16 WMMA B-fragment layout.
// Fragment (ntile, kc) = 512 bf16, lane-major 16 per lane:
//   lane -> (n = ntile*16 + (lane&15), khalf = lane>>4)
//   element i -> K = kc*32 + khalf*16 + i   (contiguous along D -> fast read)
// ---------------------------------------------------------------------------
__global__ void prep_embed_frags(const float* __restrict__ embed,
                                 __bf16* __restrict__ ebf) {
  int gid  = blockIdx.x * blockDim.x + threadIdx.x;  // [0, NTILES*KCHUNKS*32)
  int lane = gid & 31;
  int frag = gid >> 5;                 // ntile*KCHUNKS + kc
  int ntile = frag / KCHUNKS;
  int kc    = frag - ntile * KCHUNKS;
  int n  = ntile * 16 + (lane & 15);
  int k0 = kc * 32 + ((lane >> 4) << 4);
  const float* src = embed + (size_t)n * DIM + k0;
  __bf16* dst = ebf + (size_t)frag * 512 + lane * 16;
#pragma unroll
  for (int i = 0; i < 16; ++i) dst[i] = (__bf16)src[i];
}

// ---------------------------------------------------------------------------
// Kernel 2: e_sq[c] = sum(embed[c]^2); block 0 also zeroes the loss scalar.
// ---------------------------------------------------------------------------
__global__ void esq_kernel(const float* __restrict__ embed,
                           float* __restrict__ e_sq,
                           float* __restrict__ loss) {
  __shared__ float red[128];
  int c = blockIdx.x;
  float s = 0.f;
  for (int d = threadIdx.x; d < DIM; d += 128) {
    float v = embed[(size_t)c * DIM + d];
    s += v * v;
  }
  red[threadIdx.x] = s;
  __syncthreads();
  for (int off = 64; off >= 1; off >>= 1) {
    if (threadIdx.x < off) red[threadIdx.x] += red[threadIdx.x + off];
    __syncthreads();
  }
  if (threadIdx.x == 0) {
    e_sq[c] = red[0];
    if (c == 0) *loss = 0.f;
  }
}

// ---------------------------------------------------------------------------
// Kernel 3: fused bf16-WMMA GEMM (x @ embed^T) + per-token argmax of
// score = 2*dot - e_sq[n].  One wave per 16-token tile, N-blocked by 4,
// with explicit double-buffering of the 4 B fragments so ~8 global loads
// are in flight across each 4-WMMA group.
// ---------------------------------------------------------------------------
__global__ void __launch_bounds__(32 * WAVES_PER_WG)
argmax_gemm(const float* __restrict__ x, const __bf16* __restrict__ ebf,
            const float* __restrict__ e_sq, int* __restrict__ idxbuf,
            float* __restrict__ out_idx_f) {
  extern __shared__ __attribute__((aligned(32))) __bf16 alds[];  // [4][KCHUNKS][512]
  const int tid  = threadIdx.x;
  const int lane = tid & 31;
  const int w    = tid >> 5;
  const int mt   = blockIdx.x * WAVES_PER_WG + w;      // 16-token tile index
  const int row  = mt * 16 + (lane & 15);
  const int khalf = (lane >> 4) ? 16 : 0;              // A-layout K sub-base
  __bf16* my = alds + (size_t)w * (KCHUNKS * 512);

  // Stage A tile: lane<16 holds K 0-15; lane>=16 holds K 16-31.
  {
    const float* xrow = x + (size_t)row * DIM;
    for (int kc = 0; kc < KCHUNKS; ++kc) {
      const int kb = kc * 32 + khalf;
      __bf16* dst = my + kc * 512 + lane * 16;
#pragma unroll
      for (int i = 0; i < 8; ++i) dst[i] = (__bf16)xrow[kb + i];
#pragma unroll
      for (int i = 0; i < 8; ++i) dst[8 + i] = (__bf16)xrow[kb + 8 + i];
    }
  }
  __syncthreads();

  float best_val[8];
  int   best_idx[8];
#pragma unroll
  for (int r = 0; r < 8; ++r) { best_val[r] = -3.0e38f; best_idx[r] = 0; }

  const __bf16* afrag = my + lane * 16;

  for (int ntg = 0; ntg < NTILES / NBLK; ++ntg) {      // 64 groups of 4 tiles
    const int nbase = ntg * (16 * NBLK) + (lane & 15);
    float esq[NBLK];
#pragma unroll
    for (int j = 0; j < NBLK; ++j) esq[j] = e_sq[nbase + j * 16];

    v8f acc[NBLK];
#pragma unroll
    for (int j = 0; j < NBLK; ++j) acc[j] = (v8f){};

    const __bf16* bptr =
        ebf + (size_t)(ntg * NBLK * KCHUNKS) * 512 + lane * 16;

    // Prologue: preload B fragments for kc = 0.
    v16bf bcur[NBLK];
#pragma unroll
    for (int j = 0; j < NBLK; ++j)
      bcur[j] = *(const v16bf*)(bptr + (size_t)(j * KCHUNKS) * 512);

    // Steady state: issue loads for kc+1 before the 4 WMMAs of kc.
#pragma unroll 2
    for (int kc = 0; kc < KCHUNKS - 1; ++kc) {
      v16bf a = *(const v16bf*)(afrag + kc * 512);
      v16bf bnext[NBLK];
#pragma unroll
      for (int j = 0; j < NBLK; ++j)
        bnext[j] = *(const v16bf*)(bptr + (size_t)(j * KCHUNKS + kc + 1) * 512);
#pragma unroll
      for (int j = 0; j < NBLK; ++j)
        acc[j] = __builtin_amdgcn_wmma_f32_16x16x32_bf16(
            false, a, false, bcur[j], (short)0, acc[j], false, false);
#pragma unroll
      for (int j = 0; j < NBLK; ++j) bcur[j] = bnext[j];
    }
    // Epilogue: last k-chunk.
    {
      v16bf a = *(const v16bf*)(afrag + (KCHUNKS - 1) * 512);
#pragma unroll
      for (int j = 0; j < NBLK; ++j)
        acc[j] = __builtin_amdgcn_wmma_f32_16x16x32_bf16(
            false, a, false, bcur[j], (short)0, acc[j], false, false);
    }

    // C layout: lane holds column n; VGPR r = row (m=r for lanes 0-15,
    // m=r+8 for lanes 16-31).
#pragma unroll
    for (int j = 0; j < NBLK; ++j) {
      const int n = nbase + j * 16;
#pragma unroll
      for (int r = 0; r < 8; ++r) {
        float score = 2.0f * acc[j][r] - esq[j];
        if (score > best_val[r]) { best_val[r] = score; best_idx[r] = n; }
      }
    }
  }

  // Argmax reduction across each 16-lane half (offsets 1..8 stay in-half).
#pragma unroll
  for (int off = 8; off >= 1; off >>= 1) {
#pragma unroll
    for (int r = 0; r < 8; ++r) {
      float ov = __shfl_xor(best_val[r], off, 32);
      int   oi = __shfl_xor(best_idx[r], off, 32);
      if (ov > best_val[r] || (ov == best_val[r] && oi < best_idx[r])) {
        best_val[r] = ov; best_idx[r] = oi;
      }
    }
  }
  if ((lane & 15) == 0) {
    const int mbase = mt * 16 + ((lane >> 4) << 3);
#pragma unroll
    for (int r = 0; r < 8; ++r) {
      idxbuf[mbase + r]    = best_idx[r];
      out_idx_f[mbase + r] = (float)best_idx[r];
    }
  }
}

// ---------------------------------------------------------------------------
// Kernel 4: exact fp32 gather quantized = embed[idx] and commit loss.
// ---------------------------------------------------------------------------
__global__ void gather_loss(const float* __restrict__ x,
                            const float* __restrict__ embed,
                            const int* __restrict__ idxbuf,
                            float* __restrict__ outq,
                            float* __restrict__ loss) {
  __shared__ float red[256];
  const int t    = blockIdx.x;
  const int code = idxbuf[t];
  const float* e  = embed + (size_t)code * DIM;
  const float* xr = x + (size_t)t * DIM;
  float* o = outq + (size_t)t * DIM;
  float s = 0.f;
  for (int d = threadIdx.x; d < DIM; d += 256) {
    float q = e[d];
    o[d] = q;
    float df = xr[d] - q;
    s += df * df;
  }
  red[threadIdx.x] = s;
  __syncthreads();
  for (int off = 128; off >= 1; off >>= 1) {
    if (threadIdx.x < off) red[threadIdx.x] += red[threadIdx.x + off];
    __syncthreads();
  }
  if (threadIdx.x == 0)
    atomicAdd(loss, red[0] * (1.0f / ((float)N_TOKENS * (float)DIM)));
}

// ---------------------------------------------------------------------------
extern "C" void kernel_launch(void* const* d_in, const int* in_sizes, int n_in,
                              void* d_out, int out_size, void* d_ws, size_t ws_size,
                              hipStream_t stream) {
  const float* x     = (const float*)d_in[0];   // (32768, 1280) f32
  const float* embed = (const float*)d_in[1];   // (4096, 1280)  f32

  float* out      = (float*)d_out;
  float* outq     = out;                                  // quantized (N*D)
  float* out_idx  = out + (size_t)N_TOKENS * DIM;         // indices as float (N)
  float* out_loss = out_idx + N_TOKENS;                   // scalar

  // Workspace layout (~10.6 MB)
  __bf16* ebf  = (__bf16*)d_ws;                           // embed bf16 B-fragments
  float* e_sq  = (float*)(ebf + (size_t)N_CODES * DIM);   // 4096 f32
  int*   idxbuf = (int*)(e_sq + N_CODES);                 // 32768 i32

  prep_embed_frags<<<(NTILES * KCHUNKS * 32) / 256, 256, 0, stream>>>(embed, ebf);
  esq_kernel<<<N_CODES, 128, 0, stream>>>(embed, e_sq, out_loss);

  const size_t ldsBytes = (size_t)WAVES_PER_WG * KCHUNKS * 512 * sizeof(__bf16); // 160 KB
  argmax_gemm<<<MTILES / WAVES_PER_WG, 32 * WAVES_PER_WG, ldsBytes, stream>>>(
      x, ebf, e_sq, idxbuf, out_idx);

  gather_loss<<<N_TOKENS, 256, 0, stream>>>(x, embed, idxbuf, outq, out_loss);
}
